// ResourceAllocatorDecoder_47425028883085
// MI455X (gfx1250) — compile-verified, matrix operand order checked by reading, not run on hardware
//
#include <hip/hip_runtime.h>
#include <hip/hip_bf16.h>

#define B_   16
#define U_   2048
#define S_   512
#define D_   128
#define F_   32
#define KRED 3
#define ITERS 16
#define NEGV (-1000000000.0f)
#define INV_SQRT_D 0.0883883476483184f

typedef __attribute__((ext_vector_type(16))) __bf16 v16bf;
typedef __attribute__((ext_vector_type(8)))  __bf16 v8bf;
typedef __attribute__((ext_vector_type(8)))  float  v8f;
typedef __attribute__((ext_vector_type(4)))  unsigned uint4v;
typedef __attribute__((ext_vector_type(8)))  int      int8v;
typedef __attribute__((ext_vector_type(4)))  int      int4v;

#if defined(__has_builtin)
#  if __has_builtin(__builtin_amdgcn_tensor_load_to_lds) && \
      __has_builtin(__builtin_amdgcn_s_wait_tensorcnt)
#    define USE_TDM 1
#  endif
#endif
#ifndef USE_TDM
#  define USE_TDM 0
#endif

// ---------------------------------------------------------------- conversions
__global__ void conv_f2bf(const float* __restrict__ src, __bf16* __restrict__ dst, long n) {
  long i = (long)blockIdx.x * blockDim.x + threadIdx.x;
  long stride = (long)gridDim.x * blockDim.x;
  for (; i < n; i += stride) dst[i] = (__bf16)src[i];
}

// Mt[e][e2] = M[e2][e] = sum_d Wq[d][e2] * Wk[d][e]   (q2 = user_enc * Mt^T, NT-GEMM form)
__global__ void k_build_Mt(const float* __restrict__ Wq, const float* __restrict__ Wk,
                           __bf16* __restrict__ Mt) {
  int idx = blockIdx.x * blockDim.x + threadIdx.x;   // 128*128
  int e  = idx >> 7;
  int e2 = idx & 127;
  float acc = 0.f;
  for (int d = 0; d < D_; ++d) acc += Wq[d * D_ + e2] * Wk[d * D_ + e];
  Mt[idx] = (__bf16)acc;
}

// ---------------------------------------------------------------- simple WMMA NT-GEMM (for q2)
__global__ void gemm_nt_bf16_wmma(const __bf16* __restrict__ A, const __bf16* __restrict__ Bt,
                                  float* __restrict__ C, int Mdim, int Ndim, int Kdim,
                                  long sA, long sB, long sC) {
  const int lane = threadIdx.x & 31;
  const int wave = threadIdx.x >> 5;
  const int tilesN = Ndim >> 4;
  const int tile = blockIdx.x * (blockDim.x >> 5) + wave;
  const int tm = tile / tilesN, tn = tile % tilesN;
  const __bf16* Ab = A  + (size_t)blockIdx.z * sA;
  const __bf16* Bb = Bt + (size_t)blockIdx.z * sB;
  float*        Cb = C  + (size_t)blockIdx.z * sC;

  const int r16 = lane & 15;
  const int kb  = (lane >> 4) << 3;
  const size_t arow = (size_t)(tm * 16 + r16) * Kdim;
  const size_t brow = (size_t)(tn * 16 + r16) * Kdim;

  v8f acc = {};
  for (int kk = 0; kk < Kdim; kk += 32) {
    v8bf a0 = *(const v8bf*)(Ab + arow + kk + kb);
    v8bf a1 = *(const v8bf*)(Ab + arow + kk + kb + 16);
    v8bf b0 = *(const v8bf*)(Bb + brow + kk + kb);
    v8bf b1 = *(const v8bf*)(Bb + brow + kk + kb + 16);
    v16bf av = __builtin_shufflevector(a0, a1, 0,1,2,3,4,5,6,7,8,9,10,11,12,13,14,15);
    v16bf bv = __builtin_shufflevector(b0, b1, 0,1,2,3,4,5,6,7,8,9,10,11,12,13,14,15);
    acc = __builtin_amdgcn_wmma_f32_16x16x32_bf16(false, av, false, bv, (short)0, acc,
                                                  false, false);
  }
  const int rbase = tm * 16 + ((lane >> 4) << 3);
  const int col   = tn * 16 + r16;
#pragma unroll
  for (int r = 0; r < 8; ++r) Cb[(size_t)(rbase + r) * Ndim + col] = acc[r];
}

// ------------------------------------------------- main WMMA GEMM with TDM-staged A strip
// base_t[b, s, u]: A = server_enc_bf (S x K), Bt = q2_bf (U x K).
// Block: 8 waves; block covers 16 M-rows x 512 N-cols; wave covers 4 N-tiles sharing one
// A fragment (register reuse + 4 independent WMMAs per K step). A strip staged in LDS by TDM.
__global__ void gemm_g3_wmma(const __bf16* __restrict__ A, const __bf16* __restrict__ Bt,
                             float* __restrict__ C) {
  const int Kdim = D_, Ndim = U_;
  __shared__ __align__(16) __bf16 As[16 * D_];

  const int lane = threadIdx.x & 31;
  const int wave = threadIdx.x >> 5;
  const int tm = blockIdx.x >> 2;            // 32 M strips
  const int ng = blockIdx.x & 3;             // 4 N groups of 512 cols
  const __bf16* Ab = A  + (size_t)blockIdx.z * (S_ * D_);
  const __bf16* Bb = Bt + (size_t)blockIdx.z * (U_ * D_);
  float*        Cb = C  + (size_t)blockIdx.z * ((size_t)S_ * U_);

  const __bf16* Ag = Ab + (size_t)tm * 16 * Kdim;   // 16 rows x 128 K, contiguous 4KB

#if USE_TDM
  if (threadIdx.x == 0) {
    unsigned long long ga = (unsigned long long)(uintptr_t)Ag;
    unsigned ldsoff = (unsigned)(uintptr_t)&As[0];   // generic->LDS offset (addr[31:0])
    // D# group0: count=1 | lds_addr | global_addr[56:0] | type=2
    uint4v g0;
    g0.x = 1u;                                   // count = 1, user mode
    g0.y = ldsoff;                               // lds_addr
    g0.z = (unsigned)ga;                         // global_addr[31:0]
    g0.w = (unsigned)((ga >> 32) & 0x1FFFFFFull) | (2u << 30);  // ga[56:32] | type=2
    // D# group1 (bits little-endian across 8 dwords):
    int8v g1;
    g1[0] = (int)(1u << 16);                     // workgroup_mask=0, data_size=1 (2B)
    g1[1] = (int)(((unsigned)Kdim & 0xFFFFu) << 16);  // tensor_dim0[15:0] @ bits63:48
    g1[2] = (int)((16u & 0xFFFFu) << 16);        // tensor_dim0 hi=0; tensor_dim1=16
    g1[3] = (int)(((unsigned)Kdim) << 16);       // tensor_dim1 hi=0; tile_dim0=128
    g1[4] = 16;                                  // tile_dim1=16, tile_dim2=0
    g1[5] = Kdim;                                // tensor_dim0_stride[31:0]
    g1[6] = 0;                                   // stride0 hi, stride1 lo
    g1[7] = 0;
    int4v z4 = {0, 0, 0, 0};
    int8v z8 = {0, 0, 0, 0, 0, 0, 0, 0};
    // 6-arg form (clang-23 / amdgpu-toolchain): g0, g1, g2, g3, extra group, cpol
    __builtin_amdgcn_tensor_load_to_lds(g0, g1, z4, z4, z8, 0);
    __builtin_amdgcn_s_wait_tensorcnt(0);
  }
#else
  for (int i = threadIdx.x; i < 16 * D_; i += 256) As[i] = Ag[i];
#endif
  __syncthreads();

  const int r16 = lane & 15;
  const int kb  = (lane >> 4) << 3;
  const int tn0 = ng * 32 + wave * 4;         // first of 4 N tiles for this wave

  v8f acc0 = {}, acc1 = {}, acc2 = {}, acc3 = {};
  const size_t br0 = (size_t)((tn0 + 0) * 16 + r16) * Kdim;
  const size_t br1 = (size_t)((tn0 + 1) * 16 + r16) * Kdim;
  const size_t br2 = (size_t)((tn0 + 2) * 16 + r16) * Kdim;
  const size_t br3 = (size_t)((tn0 + 3) * 16 + r16) * Kdim;

#pragma unroll
  for (int kk = 0; kk < D_; kk += 32) {
    v8bf a0 = *(const v8bf*)&As[r16 * D_ + kk + kb];
    v8bf a1 = *(const v8bf*)&As[r16 * D_ + kk + kb + 16];
    v16bf av = __builtin_shufflevector(a0, a1, 0,1,2,3,4,5,6,7,8,9,10,11,12,13,14,15);

    v8bf b00 = *(const v8bf*)(Bb + br0 + kk + kb);
    v8bf b01 = *(const v8bf*)(Bb + br0 + kk + kb + 16);
    v8bf b10 = *(const v8bf*)(Bb + br1 + kk + kb);
    v8bf b11 = *(const v8bf*)(Bb + br1 + kk + kb + 16);
    v8bf b20 = *(const v8bf*)(Bb + br2 + kk + kb);
    v8bf b21 = *(const v8bf*)(Bb + br2 + kk + kb + 16);
    v8bf b30 = *(const v8bf*)(Bb + br3 + kk + kb);
    v8bf b31 = *(const v8bf*)(Bb + br3 + kk + kb + 16);
    v16bf bv0 = __builtin_shufflevector(b00, b01, 0,1,2,3,4,5,6,7,8,9,10,11,12,13,14,15);
    v16bf bv1 = __builtin_shufflevector(b10, b11, 0,1,2,3,4,5,6,7,8,9,10,11,12,13,14,15);
    v16bf bv2 = __builtin_shufflevector(b20, b21, 0,1,2,3,4,5,6,7,8,9,10,11,12,13,14,15);
    v16bf bv3 = __builtin_shufflevector(b30, b31, 0,1,2,3,4,5,6,7,8,9,10,11,12,13,14,15);

    acc0 = __builtin_amdgcn_wmma_f32_16x16x32_bf16(false, av, false, bv0, (short)0, acc0, false, false);
    acc1 = __builtin_amdgcn_wmma_f32_16x16x32_bf16(false, av, false, bv1, (short)0, acc1, false, false);
    acc2 = __builtin_amdgcn_wmma_f32_16x16x32_bf16(false, av, false, bv2, (short)0, acc2, false, false);
    acc3 = __builtin_amdgcn_wmma_f32_16x16x32_bf16(false, av, false, bv3, (short)0, acc3, false, false);
  }

  const int rbase = tm * 16 + ((lane >> 4) << 3);
#pragma unroll
  for (int r = 0; r < 8; ++r) {
    size_t rowoff = (size_t)(rbase + r) * Ndim;
    Cb[rowoff + (tn0 + 0) * 16 + r16] = acc0[r];
    Cb[rowoff + (tn0 + 1) * 16 + r16] = acc1[r];
    Cb[rowoff + (tn0 + 2) * 16 + r16] = acc2[r];
    Cb[rowoff + (tn0 + 3) * 16 + r16] = acc3[r];
  }
}

// q4[g,c] = sum_e q2[g,e]*spw[e,c]  (c<4),  qb[g] = sum_e q2[g,e]*spb[e]
__global__ void k_q4qb(const float* __restrict__ q2, const float* __restrict__ spw,
                       const float* __restrict__ spb, float* __restrict__ q4,
                       float* __restrict__ qb) {
  int g = blockIdx.x * blockDim.x + threadIdx.x;
  if (g >= B_ * U_) return;
  const float* row = q2 + (size_t)g * D_;
  float a0 = 0, a1 = 0, a2 = 0, a3 = 0, ab = 0;
  for (int e = 0; e < D_; ++e) {
    float v = row[e];
    a0 += v * spw[e * 4 + 0]; a1 += v * spw[e * 4 + 1];
    a2 += v * spw[e * 4 + 2]; a3 += v * spw[e * 4 + 3];
    ab += v * spb[e];
  }
  q4[(size_t)g * 4 + 0] = a0; q4[(size_t)g * 4 + 1] = a1;
  q4[(size_t)g * 4 + 2] = a2; q4[(size_t)g * 4 + 3] = a3;
  qb[g] = ab;
}

// ---------------------------------------------------------------- init kernels
__global__ void init_cbits(const int* __restrict__ connect, unsigned* __restrict__ cbits) {
  int w = blockIdx.x * blockDim.x + threadIdx.x;        // B*S*(U/32)
  if (w >= B_ * S_ * (U_ / 32)) return;
  int k = w % (U_ / 32);
  int bs = w / (U_ / 32);
  int b = bs / S_, s = bs % S_;
  unsigned bits = 0;
  for (int i = 0; i < 32; ++i) {
    int u = k * 32 + i;
    if (connect[((size_t)(b * U_ + u)) * S_ + s] != 0) bits |= (1u << i);
  }
  cbits[w] = bits;
}

__global__ void init_fdm(const int* __restrict__ fdoh, unsigned* __restrict__ fdm) {
  int bs = blockIdx.x * blockDim.x + threadIdx.x;
  if (bs >= B_ * S_) return;
  unsigned m = 0;
  for (int f = 0; f < F_; ++f)
    if (fdoh[(size_t)bs * F_ + f] != 0) m |= (1u << f);
  fdm[bs] = m;
}

__global__ void init_state(int* rcb, unsigned* seenb, float* logp, unsigned* abits,
                           unsigned* flags, float* cap, float* needs4,
                           const float* __restrict__ servers, const float* __restrict__ users) {
  int i = blockIdx.x * blockDim.x + threadIdx.x;        // covers B*S*(U/32)
  if (i < B_ * U_) { rcb[i] = 0; seenb[i] = 0u; logp[i] = 0.f; }
  if (i < B_ * S_ * 4) cap[i] = servers[(size_t)(i >> 2) * 7 + 3 + (i & 3)];
  if (i < B_ * U_ * 4) needs4[i] = users[(size_t)(i >> 2) * 6 + 2 + (i & 3)];
  if (i < B_ * S_ * (U_ / 32)) abits[i] = 0u;
  if (i < ITERS * 2) flags[i] = 0u;
}

// ---------------------------------------------------------------- decode iteration
// per-user terms hoisted out of the (b,s,u) inner loop
__global__ void k0_peru(const int* __restrict__ rcb, const float* __restrict__ qb,
                        const float* __restrict__ usw, const float* __restrict__ score_bias_p,
                        float* __restrict__ pu) {
  int g = blockIdx.x * blockDim.x + threadIdx.x;
  if (g >= B_ * U_) return;
  int rc = rcb[g];
  float rcf = (float)rc;
  float us = (rcf * (1.f / 3.f)) * usw[0] +
             (fmaxf(3.f - rcf, 0.f) * (1.f / 3.f)) * usw[1] +
             ((rc == KRED - 1) ? 1.f : 0.f) * usw[2];
  pu[g] = INV_SQRT_D * qb[g] + score_bias_p[0] + us;
}

// one wave per (b,s): online softmax/argmax over u, eligibility, flags
__global__ void k1_reduce(const float* __restrict__ base_t, const float* __restrict__ q4,
                          const float* __restrict__ pu, const int* __restrict__ rcb,
                          const unsigned* __restrict__ seenb, const unsigned* __restrict__ fdm,
                          const unsigned* __restrict__ cbits, const unsigned* __restrict__ abits,
                          const float* __restrict__ cap, const float* __restrict__ servers,
                          const float* __restrict__ needs4, const float* __restrict__ fd_bias_p,
                          float* cmax, int* cargu, float* lpp, unsigned* vflag,
                          unsigned* flags, int iter) {
  const int lane = threadIdx.x & 31;
  const int bs = blockIdx.x * (blockDim.x >> 5) + (threadIdx.x >> 5);
  const int b = bs / S_;

  float capv[4], ratio[4];
#pragma unroll
  for (int c = 0; c < 4; ++c) {
    capv[c] = cap[(size_t)bs * 4 + c];
    float ic = servers[(size_t)bs * 7 + 3 + c];
    ratio[c] = capv[c] / fmaxf(ic, 1e-6f);
  }
  const unsigned fm = fdm[bs];
  const float fb = fd_bias_p[0];

  float m = -3.0e38f, ssum = 0.f;
  int arg = 0;
  unsigned anyE = 0, anyN = 0;

  for (int k = 0; k < U_ / 32; ++k) {
    const int u = k * 32 + lane;
    const int g = b * U_ + u;
    const int rc = rcb[g];
    const unsigned cw = cbits[(size_t)bs * (U_ / 32) + k];
    const unsigned aw = abits[(size_t)bs * (U_ / 32) + k];
    const bool nmore = rc < KRED;
    anyN |= (unsigned)nmore;

    const float4 nd = *(const float4*)(needs4 + (size_t)g * 4);
    const bool can = (capv[0] >= nd.x) && (capv[1] >= nd.y) &&
                     (capv[2] >= nd.z) && (capv[3] >= nd.w);
    const bool el = ((cw >> lane) & 1u) && can && nmore && !((aw >> lane) & 1u);
    anyE |= (unsigned)el;

    const float4 q4v = *(const float4*)(q4 + (size_t)g * 4);
    const float q4d = q4v.x * ratio[0] + q4v.y * ratio[1] + q4v.z * ratio[2] + q4v.w * ratio[3];
    const float isnew = ((seenb[g] & fm) == 0u) ? 1.f : 0.f;
    const float logit = INV_SQRT_D * (base_t[(size_t)bs * U_ + u] + q4d) + pu[g] + fb * isnew;
    const float cand = el ? logit : NEGV;

    const float nm2 = fmaxf(m, cand);
    ssum = ssum * __expf(m - nm2) + __expf(cand - nm2);
    if (cand > m) arg = u;
    m = nm2;
  }

  for (int off = 16; off; off >>= 1) {
    float om = __shfl_xor(m, off, 32);
    float os = __shfl_xor(ssum, off, 32);
    int   oa = __shfl_xor(arg, off, 32);
    unsigned oe = __shfl_xor(anyE, off, 32);
    unsigned on = __shfl_xor(anyN, off, 32);
    float nm2 = fmaxf(m, om);
    float ns = ssum * __expf(m - nm2) + os * __expf(om - nm2);
    if (om > m || (om == m && oa < arg)) arg = oa;
    m = nm2; ssum = ns; anyE |= oe; anyN |= on;
  }

  if (lane == 0) {
    cmax[bs] = m; cargu[bs] = arg; lpp[bs] = -__logf(ssum); vflag[bs] = anyE;
    if (anyE) atomicOr(&flags[iter * 2 + 0], 1u);
    if (anyN) atomicOr(&flags[iter * 2 + 1], 1u);
  }
}

__global__ void k_go(unsigned* flags, unsigned* go, int iter) {
  unsigned prev = (iter == 0) ? 1u : go[iter - 1];
  go[iter] = prev & flags[iter * 2 + 0] & flags[iter * 2 + 1];
}

__global__ void k2a_reset(unsigned long long* slots, unsigned* sgain) {
  int g = blockIdx.x * blockDim.x + threadIdx.x;
  if (g < B_ * U_) { slots[g] = 0ull; sgain[g] = 0u; }
}

__device__ inline unsigned fordu(float f) {
  unsigned u = __float_as_uint(f);
  return (u & 0x80000000u) ? ~u : (u | 0x80000000u);
}

__global__ void k2b_scatter(const float* __restrict__ cmax, const int* __restrict__ cargu,
                            const unsigned* __restrict__ vflag, const unsigned* __restrict__ fdm,
                            unsigned long long* slots, unsigned* sgain) {
  int bs = blockIdx.x * blockDim.x + threadIdx.x;
  if (bs >= B_ * S_) return;
  if (!vflag[bs]) return;
  int b = bs / S_, s = bs % S_;
  int u = cargu[bs];
  unsigned long long key =
      ((unsigned long long)fordu(cmax[bs]) << 32) | (unsigned long long)(unsigned)(S_ - 1 - s);
  atomicMax(&slots[(size_t)b * U_ + u], key);
  atomicOr(&sgain[(size_t)b * U_ + u], fdm[bs]);
}

__global__ void k3_apply(const unsigned* __restrict__ go, int iter,
                         const unsigned long long* __restrict__ slots,
                         const unsigned* __restrict__ sgain, const float* __restrict__ lpp,
                         const float* __restrict__ needs4, int* rcb, unsigned* seenb,
                         unsigned* abits, float* cap, float* logp) {
  if (!go[iter]) return;
  int g = blockIdx.x * blockDim.x + threadIdx.x;
  if (g >= B_ * U_) return;
  int b = g / U_, u = g % U_;
  seenb[g] |= sgain[g];
  unsigned long long sl = slots[g];
  if (sl) {
    int s = (S_ - 1) - (int)(unsigned)(sl & 0xffffffffull);
    logp[g] += lpp[b * S_ + s];
    rcb[g] += 1;
    size_t w = (size_t)(b * S_ + s) * (U_ / 32) + (u >> 5);
    abits[w] |= (1u << (u & 31));            // each server accepted by <=1 user: race-free
    const float4 nd = *(const float4*)(needs4 + (size_t)g * 4);
    size_t ci = (size_t)(b * S_ + s) * 4;
    cap[ci + 0] = fmaxf(cap[ci + 0] - nd.x, 0.f);
    cap[ci + 1] = fmaxf(cap[ci + 1] - nd.y, 0.f);
    cap[ci + 2] = fmaxf(cap[ci + 2] - nd.z, 0.f);
    cap[ci + 3] = fmaxf(cap[ci + 3] - nd.w, 0.f);
  }
}

// ---------------------------------------------------------------- finalize
__global__ void fin_logp(const float* __restrict__ logp, float* out) {
  __shared__ float sm[256];
  int b = blockIdx.x;
  float acc = 0.f;
  for (int u = threadIdx.x; u < U_; u += 256) acc += logp[b * U_ + u];
  sm[threadIdx.x] = acc;
  __syncthreads();
  for (int o = 128; o; o >>= 1) {
    if ((int)threadIdx.x < o) sm[threadIdx.x] += sm[threadIdx.x + o];
    __syncthreads();
  }
  if (threadIdx.x == 0) out[b] = sm[0];
}

__global__ void fin_alloc(const unsigned* __restrict__ abits, float* __restrict__ out) {
  size_t n = (size_t)B_ * U_ * S_;
  for (size_t i = (size_t)blockIdx.x * blockDim.x + threadIdx.x; i < n;
       i += (size_t)gridDim.x * blockDim.x) {
    int b = (int)(i / ((size_t)U_ * S_));
    size_t r = i % ((size_t)U_ * S_);
    int u = (int)(r / S_);
    int s = (int)(r % S_);
    unsigned w = abits[(size_t)(b * S_ + s) * (U_ / 32) + (u >> 5)];
    out[i] = ((w >> (u & 31)) & 1u) ? 1.f : 0.f;
  }
}

// ---------------------------------------------------------------- host
extern "C" void kernel_launch(void* const* d_in, const int* in_sizes, int n_in,
                              void* d_out, int out_size, void* d_ws, size_t ws_size,
                              hipStream_t stream) {
  (void)in_sizes; (void)n_in; (void)out_size; (void)ws_size;
  const float* user_enc   = (const float*)d_in[0];
  const float* server_enc = (const float*)d_in[1];
  const float* users      = (const float*)d_in[2];
  const float* servers    = (const float*)d_in[3];
  const int*   connect    = (const int*)d_in[4];
  const int*   fdoh       = (const int*)d_in[5];
  const float* spw        = (const float*)d_in[6];
  const float* spb        = (const float*)d_in[7];
  const float* fd_bias    = (const float*)d_in[8];
  const float* Wq         = (const float*)d_in[9];
  const float* Wk         = (const float*)d_in[10];
  const float* usw        = (const float*)d_in[11];
  const float* score_bias = (const float*)d_in[12];

  float* out_logp  = (float*)d_out;
  float* out_alloc = out_logp + B_;
  float* out_cap   = out_alloc + (size_t)B_ * U_ * S_;

  char* ws = (char*)d_ws;
  size_t off = 0;
  auto take = [&](size_t bytes) -> void* {
    void* p = ws + off;
    off = (off + bytes + 255) & ~(size_t)255;
    return p;
  };
  float*    base_t = (float*)take((size_t)B_ * S_ * U_ * 4);
  float*    q2f    = (float*)take((size_t)B_ * U_ * D_ * 4);
  __bf16*   ue_bf  = (__bf16*)take((size_t)B_ * U_ * D_ * 2);
  __bf16*   q2bf   = (__bf16*)take((size_t)B_ * U_ * D_ * 2);
  __bf16*   se_bf  = (__bf16*)take((size_t)B_ * S_ * D_ * 2);
  __bf16*   Mt_bf  = (__bf16*)take((size_t)D_ * D_ * 2);
  float*    q4     = (float*)take((size_t)B_ * U_ * 4 * 4);
  float*    qb     = (float*)take((size_t)B_ * U_ * 4);
  float*    pu     = (float*)take((size_t)B_ * U_ * 4);
  float*    needs4 = (float*)take((size_t)B_ * U_ * 4 * 4);
  int*      rcb    = (int*)take((size_t)B_ * U_ * 4);
  unsigned* seenb  = (unsigned*)take((size_t)B_ * U_ * 4);
  unsigned* fdm    = (unsigned*)take((size_t)B_ * S_ * 4);
  unsigned* cbits  = (unsigned*)take((size_t)B_ * S_ * (U_ / 32) * 4);
  unsigned* abits  = (unsigned*)take((size_t)B_ * S_ * (U_ / 32) * 4);
  float*    logp   = (float*)take((size_t)B_ * U_ * 4);
  float*    cmax   = (float*)take((size_t)B_ * S_ * 4);
  int*      cargu  = (int*)take((size_t)B_ * S_ * 4);
  float*    lpp    = (float*)take((size_t)B_ * S_ * 4);
  unsigned* vflag  = (unsigned*)take((size_t)B_ * S_ * 4);
  unsigned long long* slots = (unsigned long long*)take((size_t)B_ * U_ * 8);
  unsigned* sgain  = (unsigned*)take((size_t)B_ * U_ * 4);
  unsigned* flags  = (unsigned*)take(ITERS * 2 * 4);
  unsigned* go     = (unsigned*)take(ITERS * 4);

  // precompute: conversions, Wq^T@Wk, q2, q4/qb, base
  conv_f2bf<<<2048, 256, 0, stream>>>(user_enc, ue_bf, (long)B_ * U_ * D_);
  conv_f2bf<<<1024, 256, 0, stream>>>(server_enc, se_bf, (long)B_ * S_ * D_);
  k_build_Mt<<<64, 256, 0, stream>>>(Wq, Wk, Mt_bf);
  gemm_nt_bf16_wmma<<<dim3(4096, 1, 1), 128, 0, stream>>>(
      ue_bf, Mt_bf, q2f, B_ * U_, D_, D_, 0, 0, 0);
  k_q4qb<<<(B_ * U_) / 256, 256, 0, stream>>>(q2f, spw, spb, q4, qb);
  conv_f2bf<<<2048, 256, 0, stream>>>(q2f, q2bf, (long)B_ * U_ * D_);
  gemm_g3_wmma<<<dim3(128, 1, B_), 256, 0, stream>>>(se_bf, q2bf, base_t);

  // state init (re-done every call; no cross-call state)
  init_cbits<<<(B_ * S_ * (U_ / 32)) / 256, 256, 0, stream>>>(connect, cbits);
  init_fdm<<<(B_ * S_) / 256, 256, 0, stream>>>(fdoh, fdm);
  init_state<<<(B_ * S_ * (U_ / 32)) / 256, 256, 0, stream>>>(
      rcb, seenb, logp, abits, flags, out_cap, needs4, servers, users);

  for (int it = 0; it < ITERS; ++it) {
    k0_peru<<<(B_ * U_) / 256, 256, 0, stream>>>(rcb, qb, usw, score_bias, pu);
    k1_reduce<<<(B_ * S_) / 8, 256, 0, stream>>>(
        base_t, q4, pu, rcb, seenb, fdm, cbits, abits, out_cap, servers, needs4,
        fd_bias, cmax, cargu, lpp, vflag, flags, it);
    k_go<<<1, 1, 0, stream>>>(flags, go, it);
    k2a_reset<<<(B_ * U_) / 256, 256, 0, stream>>>(slots, sgain);
    k2b_scatter<<<(B_ * S_) / 256, 256, 0, stream>>>(cmax, cargu, vflag, fdm, slots, sgain);
    k3_apply<<<(B_ * U_) / 256, 256, 0, stream>>>(go, it, slots, sgain, lpp, needs4, rcb,
                                                  seenb, abits, out_cap, logp);
  }

  fin_logp<<<B_, 256, 0, stream>>>(logp, out_logp);
  fin_alloc<<<8192, 256, 0, stream>>>(abits, out_alloc);
}
